// SPE_69191923138592
// MI455X (gfx1250) — matrix operand order repaired and checked by reading, not compiled, not run on hardware
//
#include <hip/hip_runtime.h>
#include <hip/hip_bf16.h>

typedef __bf16 bf16;
typedef __attribute__((ext_vector_type(16))) __bf16 v16bf;
typedef __attribute__((ext_vector_type(8)))  __bf16 v8bf;
typedef __attribute__((ext_vector_type(8)))  float  v8f;
typedef __attribute__((ext_vector_type(2)))  float  v2f;

#define BB   16
#define CC   256
#define MIDC 128
#define HH   128
#define WW   128
#define HWP  (HH * WW)          // 16384
#define HP   130
#define WP   130

__device__ __forceinline__ float leaky(float v) { return v >= 0.0f ? v : 0.2f * v; }

// ---------------- workspace layout (bytes), all offsets 256B-aligned ----------------
static constexpr size_t OFF_XPAD   = 0;                        // bf16 [B][130][130][C]
static constexpr size_t OFF_H1     = 138444800;                // bf16 [B][128][128][MID]
static constexpr size_t OFF_XC     = OFF_H1 + 67108864;        // f32  [B][16384][C]
static constexpr size_t OFF_POOLED = OFF_XC + 268435456;       // f32  [B][C]
static constexpr size_t OFF_SCORE  = OFF_POOLED + 16384;       // f32  [B][C][C]
static constexpr size_t OFF_W1P    = OFF_SCORE + 4194304;      // bf16 packed fragments
static constexpr size_t OFF_W2P    = OFF_W1P + 589824;         // bf16 packed fragments

// ---------------- init: zero pooled accumulator ----------------
__global__ void k_init(float* pooled) {
    int i = blockIdx.x * 256 + threadIdx.x;
    if (i < BB * CC) pooled[i] = 0.0f;
}

// ---------------- pack conv weights f32 -> bf16 in WMMA A-fragment order ----------------
// w1p: [ocTile(8)][icb(8)][tap(9)][lane(32)][e(16)]   (A 16x32 bf16 layout)
// w2p: [ocTile(16)][kc(4)][lane(32)][e(16)]
__global__ void k_pack_w(const float* __restrict__ w1, const float* __restrict__ w2,
                         bf16* __restrict__ w1p, bf16* __restrict__ w2p) {
    int i = blockIdx.x * 256 + threadIdx.x;
    const int N1 = MIDC * CC * 9;     // 294912
    const int N2 = CC * MIDC;         // 32768
    if (i < N1) {
        int e = i & 15, lane = (i >> 4) & 31;
        int t = i >> 9;               // (ocTile*8 + icb)*9 + tap
        int tap = t % 9, q = t / 9;
        int icb = q & 7, ocTile = q >> 3;
        int n = lane & 15, g = lane >> 4;
        int K = e + 16 * g;
        w1p[i] = (bf16)w1[((long)(ocTile * 16 + n) * CC + icb * 32 + K) * 9 + tap];
    } else if (i < N1 + N2) {
        int j = i - N1;
        int e = j & 15, lane = (j >> 4) & 31;
        int t = j >> 9;               // ocTile*4 + kc
        int kc = t & 3, ocTile = t >> 2;
        int n = lane & 15, g = lane >> 4;
        int K = e + 16 * g;
        w2p[j] = (bf16)w2[(ocTile * 16 + n) * MIDC + kc * 32 + K];
    }
}

// ---------------- reflect-pad + f32->bf16 + NCHW -> channel-last (LDS transpose) --------
__global__ __launch_bounds__(256) void k_pad(const float* __restrict__ x,
                                             bf16* __restrict__ xpad) {
    __shared__ __align__(16) bf16 sT[65][32];
    int blk = blockIdx.x;
    int ctile = blk & 7, half = (blk >> 3) & 1;
    int r = blk >> 4;                  // b*130 + yp
    int yp = r % HP, b = r / HP;
    int c0 = ctile * 32, xp0 = half * 65;
    int sy = yp - 1; sy = (sy < 0) ? 1 : (sy >= HH ? HH - 2 : sy);   // jnp reflect
    int tid = threadIdx.x;
    for (int e = tid; e < 32 * 65; e += 256) {           // coalesced along x
        int c = e / 65, px = e % 65;
        int sx = xp0 + px - 1; sx = (sx < 0) ? 1 : (sx >= WW ? WW - 2 : sx);
        sT[px][c] = (bf16)x[(((long)b * CC + c0 + c) * HH + sy) * WW + sx];
    }
    __syncthreads();
    for (int ch = tid; ch < 65 * 4; ch += 256) {         // coalesced 16B channel chunks
        int px = ch >> 2, c8 = (ch & 3) * 8;
        bf16* dst = xpad + ((long)(b * HP + yp) * WP + xp0 + px) * CC + c0 + c8;
        *(v8bf*)dst = *(const v8bf*)&sT[px][c8];
    }
}

// ---------------- conv1: 3x3 reflect conv 256->128 + leaky (bf16 WMMA) ----------------
// 64-px blocks: each wave owns one oc-tile and 4 pixel-subtiles; every A fragment
// (2x global b128) feeds 4 WMMAs -> 4x less L2 weight traffic.
__global__ __launch_bounds__(256) void k_conv1(const bf16* __restrict__ xpad,
                                               const bf16* __restrict__ w1p,
                                               bf16* __restrict__ h1t) {
    __shared__ __align__(16) bf16 sIn[3][66][40];  // [row][col 0..65][ic pad 40]
    int tid = threadIdx.x, lane = tid & 31, wv = tid >> 5;
    int blk = blockIdx.x;
    int xt = blk & 1, y = (blk >> 1) & 127, b = blk >> 8;
    int x0 = xt * 64;
    int n = lane & 15, g = lane >> 4;
    const bf16* wbase = w1p + (long)wv * (8 * 9 * 32 * 16) + lane * 16;
    v8f acc0 = {}, acc1 = {}, acc2 = {}, acc3 = {};

    for (int icb = 0; icb < 8; ++icb) {
        int icBase = icb * 32;
        __syncthreads();
        for (int ch = tid; ch < 3 * 66 * 4; ch += 256) {  // 3 rows x 66 cols x 4 ic-octets
            int icc = ch & 3, cr = ch >> 2;
            int rr = cr / 66, col = cr % 66;
            const bf16* src = xpad + ((long)(b * HP + y + rr) * WP + x0 + col) * CC
                            + icBase + icc * 8;
            *(v8bf*)&sIn[rr][col][icc * 8] = *(const v8bf*)src;
        }
        if (icb < 7)
            __builtin_prefetch(xpad + ((long)(b * HP + y) * WP + x0) * CC + icBase + 32, 0, 3);
        __syncthreads();
        const bf16* wchunk = wbase + (long)icb * (9 * 32 * 16);
        #pragma unroll
        for (int tap = 0; tap < 9; ++tap) {
            int kh = tap / 3, kw = tap % 3;
            v16bf a = *(const v16bf*)(wchunk + tap * (32 * 16));      // loaded once
            union { v16bf v; v8bf h[2]; } u;
            #pragma unroll
            for (int s = 0; s < 4; ++s) {                              // reused 4x
                const bf16* bp = &sIn[kh][kw + 16 * s + n][g * 16];
                u.h[0] = *(const v8bf*)bp;
                u.h[1] = *(const v8bf*)(bp + 8);
                v8f* pa = (s == 0) ? &acc0 : (s == 1) ? &acc1 : (s == 2) ? &acc2 : &acc3;
                *pa = __builtin_amdgcn_wmma_f32_16x16x32_bf16(false, a, false, u.v,
                                                              (short)0, *pa, false, false);
            }
        }
    }
    #pragma unroll
    for (int s = 0; s < 4; ++s) {
        v8f a = (s == 0) ? acc0 : (s == 1) ? acc1 : (s == 2) ? acc2 : acc3;
        v8bf hv;
        #pragma unroll
        for (int r = 0; r < 8; ++r) hv[r] = (bf16)leaky(a[r]);
        long pix = (long)(b * HH + y) * WW + x0 + 16 * s + n;
        *(v8bf*)(h1t + pix * MIDC + wv * 16 + g * 8) = hv;   // 8 consecutive oc, 16B store
    }
}

// ---------------- conv2: 1x1 conv 128->256 + leaky + fused avg-pool partials ----------
// 64-px blocks; wave: 2 oc-tiles x 4 px-subtiles = 8 accumulators; A loaded once per kc.
__global__ __launch_bounds__(256) void k_conv2(const bf16* __restrict__ h1t,
                                               const bf16* __restrict__ w2p,
                                               float* __restrict__ xct,
                                               float* __restrict__ pooled) {
    __shared__ __align__(16) bf16 sH[64][136];     // [px][ic], pad 136 -> conflict-free
    int tid = threadIdx.x, lane = tid & 31, wv = tid >> 5;
    int blk = blockIdx.x;
    int xt = blk & 1, y = (blk >> 1) & 127, b = blk >> 8;
    int x0 = xt * 64;
    int n = lane & 15, g = lane >> 4;
    for (int ch = tid; ch < 64 * 16; ch += 256) {  // 64 px x 16 ic-octets, 16B each side
        int px = ch >> 4, icc = (ch & 15) * 8;
        const bf16* src = h1t + ((long)(b * HH + y) * WW + x0 + px) * MIDC + icc;
        *(v8bf*)&sH[px][icc] = *(const v8bf*)src;
    }
    __syncthreads();

    const bf16* w0  = w2p + (long)(wv * 2 + 0) * (4 * 32 * 16) + lane * 16;
    const bf16* w1q = w2p + (long)(wv * 2 + 1) * (4 * 32 * 16) + lane * 16;
    v8f acc[2][4] = {};
    #pragma unroll
    for (int kc = 0; kc < 4; ++kc) {
        v16bf a0 = *(const v16bf*)(w0  + kc * (32 * 16));
        v16bf a1 = *(const v16bf*)(w1q + kc * (32 * 16));
        #pragma unroll
        for (int s = 0; s < 4; ++s) {
            union { v16bf v; v8bf h[2]; } u;
            const bf16* bp = &sH[16 * s + n][kc * 32 + g * 16];
            u.h[0] = *(const v8bf*)bp;
            u.h[1] = *(const v8bf*)(bp + 8);
            acc[0][s] = __builtin_amdgcn_wmma_f32_16x16x32_bf16(false, a0, false, u.v,
                                                               (short)0, acc[0][s], false, false);
            acc[1][s] = __builtin_amdgcn_wmma_f32_16x16x32_bf16(false, a1, false, u.v,
                                                               (short)0, acc[1][s], false, false);
        }
    }

    #pragma unroll
    for (int t = 0; t < 2; ++t) {
        int ocB = (wv * 2 + t) * 16 + g * 8;       // 8 consecutive channels per lane
        float psum[8];
        #pragma unroll
        for (int r = 0; r < 8; ++r) psum[r] = 0.0f;
        #pragma unroll
        for (int s = 0; s < 4; ++s) {
            v8f vout;
            #pragma unroll
            for (int r = 0; r < 8; ++r) {
                vout[r] = leaky(acc[t][s][r]);
                psum[r] += vout[r];
            }
            long pix = (long)(b * HH + y) * WW + x0 + 16 * s + n;
            *(v8f*)(xct + pix * CC + ocB) = vout;   // 32B channel-last store
        }
        #pragma unroll
        for (int r = 0; r < 8; ++r) {               // avg-pool partials over 64 px
            float s = psum[r];
            s += __shfl_xor(s, 1, 32);
            s += __shfl_xor(s, 2, 32);
            s += __shfl_xor(s, 4, 32);
            s += __shfl_xor(s, 8, 32);
            if (n == 0) atomicAdd(&pooled[b * CC + ocB + r], s);
        }
    }
}

// ---------------- MLP + outer + softmax -> score[B][C][C] ----------------
__global__ __launch_bounds__(256) void k_mlp(const float* __restrict__ pooled,
                                             const float* __restrict__ w1, const float* __restrict__ b1,
                                             const float* __restrict__ w2, const float* __restrict__ b2,
                                             float* __restrict__ score) {
    __shared__ float sp[CC], sh[CC / 4], sm[CC];
    int b = blockIdx.x, t = threadIdx.x;
    sp[t] = pooled[b * CC + t] * (1.0f / (float)HWP);
    __syncthreads();
    if (t < CC / 4) {
        float acc = b1[t];
        for (int c = 0; c < CC; ++c) acc += sp[c] * w1[t * CC + c];
        sh[t] = leaky(acc);
    }
    __syncthreads();
    {
        float acc = b2[t];
        for (int j = 0; j < CC / 4; ++j) acc += sh[j] * w2[t * (CC / 4) + j];
        sm[t] = acc;
    }
    __syncthreads();
    float mc = sm[t];
    float mx = -3.402823466e+38f;
    for (int e = 0; e < CC; ++e) mx = fmaxf(mx, mc * sm[e]);
    float sum = 0.0f;
    for (int e = 0; e < CC; ++e) sum += __expf(mc * sm[e] - mx);
    float inv = 1.0f / sum;
    float* row = score + ((long)b * CC + t) * CC;
    for (int e = 0; e < CC; ++e) row[e] = __expf(mc * sm[e] - mx) * inv;
}

// ---------------- attention GEMM (f32 WMMA 16x16x4) + residual ----------------
// 64-px blocks; wave: 2 c-tiles x 4 px-subtiles; each score fragment feeds 4 WMMAs.
__global__ __launch_bounds__(256) void k_attn(const float* __restrict__ xct,
                                              const float* __restrict__ score,
                                              float* __restrict__ out) {
    __shared__ __align__(16) float sX[64][260];    // [px][e], pad 260 -> aligned+conflict-free
    int tid = threadIdx.x, lane = tid & 31, wv = tid >> 5;
    int blk = blockIdx.x;
    int pt = blk & 255, b = blk >> 8;
    int p0 = pt * 64;
    for (int ch = tid; ch < 64 * 64; ch += 256) {  // 64 px x 64 float4s
        int px = ch >> 6, icc = (ch & 63) * 4;
        *(float4*)&sX[px][icc] =
            *(const float4*)(xct + ((long)b * HWP + p0 + px) * CC + icc);
    }
    __syncthreads();

    int n = lane & 15, g = lane >> 4;
    const float* sb = score + (long)b * CC * CC;
    #pragma unroll
    for (int tile = 0; tile < 2; ++tile) {
        int cBase = (wv * 2 + tile) * 16;
        const float* arow = sb + (long)(cBase + n) * CC + 2 * g;
        v8f acc0 = {}, acc1 = {}, acc2 = {}, acc3 = {};
        #pragma unroll 4
        for (int k = 0; k < CC / 4; ++k) {
            v2f a = *(const v2f*)(arow + k * 4);               // 8B global, loaded once
            #pragma unroll
            for (int s = 0; s < 4; ++s) {                      // reused 4x
                v2f bm = *(const v2f*)&sX[16 * s + n][k * 4 + 2 * g];
                v8f* pa = (s == 0) ? &acc0 : (s == 1) ? &acc1 : (s == 2) ? &acc2 : &acc3;
                *pa = __builtin_amdgcn_wmma_f32_16x16x4_f32(false, a, false, bm,
                                                            (short)0, *pa, false, false);
            }
        }
        #pragma unroll
        for (int s = 0; s < 4; ++s) {
            v8f a = (s == 0) ? acc0 : (s == 1) ? acc1 : (s == 2) ? acc2 : acc3;
            float* outp = out + ((long)b * CC + cBase) * HWP + p0 + 16 * s + n;
            #pragma unroll
            for (int r = 0; r < 8; ++r) {
                int cc = 8 * g + r;
                outp[(long)cc * HWP] = a[r] + sX[16 * s + n][cBase + cc];  // + residual
            }
        }
    }
}

extern "C" void kernel_launch(void* const* d_in, const int* in_sizes, int n_in,
                              void* d_out, int out_size, void* d_ws, size_t ws_size,
                              hipStream_t stream) {
    const float* x   = (const float*)d_in[0];
    const float* w1  = (const float*)d_in[1];
    const float* w2  = (const float*)d_in[2];
    const float* m1w = (const float*)d_in[3];
    const float* m1b = (const float*)d_in[4];
    const float* m2w = (const float*)d_in[5];
    const float* m2b = (const float*)d_in[6];
    float* out = (float*)d_out;

    char* ws = (char*)d_ws;
    bf16*  xpad   = (bf16*)(ws + OFF_XPAD);
    bf16*  h1t    = (bf16*)(ws + OFF_H1);
    float* xct    = (float*)(ws + OFF_XC);
    float* pooled = (float*)(ws + OFF_POOLED);
    float* score  = (float*)(ws + OFF_SCORE);
    bf16*  w1p    = (bf16*)(ws + OFF_W1P);
    bf16*  w2p    = (bf16*)(ws + OFF_W2P);

    k_init<<<(BB * CC + 255) / 256, 256, 0, stream>>>(pooled);
    k_pack_w<<<(MIDC * CC * 9 + CC * MIDC + 255) / 256, 256, 0, stream>>>(w1, w2, w1p, w2p);
    k_pad<<<BB * HP * 2 * 8, 256, 0, stream>>>(x, xpad);
    k_conv1<<<BB * HH * (WW / 64), 256, 0, stream>>>(xpad, w1p, h1t);
    k_conv2<<<BB * HH * (WW / 64), 256, 0, stream>>>(h1t, w2p, xct, pooled);
    k_mlp<<<BB, 256, 0, stream>>>(pooled, m1w, m1b, m2w, m2b, score);
    k_attn<<<BB * (HWP / 64), 256, 0, stream>>>(xct, score, out);
}
